// GraphConvolution_15564961481469
// MI455X (gfx1250) — compile-verified, matrix operand order checked by reading, not run on hardware
//
#include <hip/hip_runtime.h>

typedef __attribute__((ext_vector_type(16))) __bf16 v16bf;
typedef __attribute__((ext_vector_type(8)))  __bf16 v8bf;
typedef __attribute__((ext_vector_type(4)))  __bf16 v4bf;
typedef __attribute__((ext_vector_type(8)))  float  v8f;

#define NN 8192          // nodes
#define F  256           // features in/out
#define KT 32            // WMMA K chunk
#define MT 64            // rows per block in main GEMM

// Build a 16xbf16 WMMA A/B fragment from a row-major buffer (ld = KT or F):
// halves 0..7 = K[koff..koff+7], halves 8..15 = K[koff+16..koff+23]
__device__ __forceinline__ v16bf ld_frag(const __bf16* p) {
    v8bf lo = *(const v8bf*)(p);
    v8bf hi = *(const v8bf*)(p + 16);
    return __builtin_shufflevector(lo, hi, 0,1,2,3,4,5,6,7,8,9,10,11,12,13,14,15);
}

__device__ __forceinline__ v8f wmma_bf16(v16bf a, v16bf b, v8f c) {
    return __builtin_amdgcn_wmma_f32_16x16x32_bf16(
        false, a, false, b, (short)0, c, false, false);
}

// ---------------------------------------------------------------- rowsum ----
__global__ __launch_bounds__(256) void k_rowsum(const float* __restrict__ adj,
                                                float* __restrict__ dis) {
    __shared__ float red[8];
    const int row = blockIdx.x;
    const float4* p = (const float4*)(adj + (size_t)row * NN);
    float s = 0.f;
    for (int i = threadIdx.x; i < NN / 4; i += 256) {
        float4 v = p[i];
        s += v.x + v.y + v.z + v.w;
    }
    #pragma unroll
    for (int off = 16; off > 0; off >>= 1) s += __shfl_down(s, off, 32);
    if ((threadIdx.x & 31) == 0) red[threadIdx.x >> 5] = s;
    __syncthreads();
    if (threadIdx.x == 0) {
        float t = 0.f;
        #pragma unroll
        for (int i = 0; i < 8; ++i) t += red[i];
        dis[row] = (t > 0.f) ? rsqrtf(t) : 0.f;
    }
}

// ------------------------------------------------ transpose+convert x ------
__global__ __launch_bounds__(256) void k_xT(const float* __restrict__ x,
                                            __bf16* __restrict__ xT) {
    __shared__ __bf16 t[64 * 72];
    const int r0 = blockIdx.x * 64;
    const int c0 = blockIdx.y * 64;
    #pragma unroll
    for (int i = 0; i < 16; ++i) {
        int e  = threadIdx.x + i * 256;
        int lr = e >> 6, lc = e & 63;
        t[lr * 72 + lc] = (__bf16)x[(size_t)(r0 + lr) * F + (c0 + lc)];
    }
    __syncthreads();
    #pragma unroll
    for (int i = 0; i < 16; ++i) {
        int e = threadIdx.x + i * 256;
        int j = e >> 6, k = e & 63;
        xT[(size_t)(c0 + j) * NN + (r0 + k)] = t[k * 72 + j];
    }
}

// ------------------------------------------------------- convert W ---------
__global__ __launch_bounds__(256) void k_cvtW(const float* __restrict__ W,
                                              __bf16* __restrict__ Wb) {
    int i = blockIdx.x * 256 + threadIdx.x;
    Wb[i] = (__bf16)W[i];
}

// ----------------------------------------- main GEMM: h = (D A D) @ x ------
// Block: 128 threads = 4 waves; tile 64(M) x 256(N); K chunks of 32.
// Double-buffered LDS, global loads pipelined one chunk ahead.
// Each wave: 4 row-tiles x 4 col-tiles (cols wave*64..+63) = 16 WMMAs/chunk,
// fed by only 8 fragment loads (4 A + 4 B).
__global__ __launch_bounds__(128) void k_gemm_adj(const float*  __restrict__ adj,
                                                  const float*  __restrict__ dis,
                                                  const __bf16* __restrict__ xT,
                                                  __bf16*       __restrict__ hb) {
    __shared__ __bf16 lds_a[2][MT * KT];   // 2 x 4 KB
    __shared__ __bf16 lds_b[2][F  * KT];   // 2 x 16 KB

    const int tid  = threadIdx.x;
    const int wave = tid >> 5;
    const int lane = tid & 31;
    const int m0   = blockIdx.x * MT;

    const int fr   = lane & 15;
    const int koff = (lane >> 4) * 8;

    // A staging: thread covers rows sr+{0,16,32,48}, cols sc..sc+3
    const int sr = tid >> 3;
    const int sc = (tid & 7) * 4;
    // B staging: thread covers xT rows bn+{0,32,...,224}, halves bsub..bsub+7
    const int bn   = tid >> 2;
    const int bsub = (tid & 3) * 8;

    float dr[4];
    #pragma unroll
    for (int i = 0; i < 4; ++i) dr[i] = dis[m0 + sr + i * 16];

    float4 sa[4]; float4 sdc; v8bf sb[8];

    auto load_stage = [&](int kk) {
        sdc = *(const float4*)(dis + kk + sc);
        #pragma unroll
        for (int i = 0; i < 4; ++i)
            sa[i] = *(const float4*)(adj + (size_t)(m0 + sr + i * 16) * NN + kk + sc);
        #pragma unroll
        for (int i = 0; i < 8; ++i)
            sb[i] = *(const v8bf*)(xT + (size_t)(bn + i * 32) * NN + kk + bsub);
    };
    auto store_stage = [&](int buf) {
        #pragma unroll
        for (int i = 0; i < 4; ++i) {
            v4bf pk = { (__bf16)(dr[i] * sdc.x * sa[i].x),
                        (__bf16)(dr[i] * sdc.y * sa[i].y),
                        (__bf16)(dr[i] * sdc.z * sa[i].z),
                        (__bf16)(dr[i] * sdc.w * sa[i].w) };
            *(v4bf*)&lds_a[buf][(sr + i * 16) * KT + sc] = pk;
        }
        #pragma unroll
        for (int i = 0; i < 8; ++i)
            *(v8bf*)&lds_b[buf][(bn + i * 32) * KT + bsub] = sb[i];
    };

    v8f acc[16];
    #pragma unroll
    for (int i = 0; i < 16; ++i) acc[i] = (v8f){0.f,0.f,0.f,0.f,0.f,0.f,0.f,0.f};

    load_stage(0);
    store_stage(0);
    __syncthreads();

    for (int k0 = 0; k0 < NN; k0 += KT) {
        const int  cur  = (k0 >> 5) & 1;
        const bool more = (k0 + KT) < NN;
        if (more) load_stage(k0 + KT);          // global loads in flight over compute

        v16bf a[4];
        #pragma unroll
        for (int rt = 0; rt < 4; ++rt)
            a[rt] = ld_frag(&lds_a[cur][(rt * 16 + fr) * KT + koff]);
        #pragma unroll
        for (int ct = 0; ct < 4; ++ct) {
            v16bf b = ld_frag(&lds_b[cur][(wave * 64 + ct * 16 + fr) * KT + koff]);
            #pragma unroll
            for (int rt = 0; rt < 4; ++rt)
                acc[ct * 4 + rt] = wmma_bf16(a[rt], b, acc[ct * 4 + rt]);
        }

        if (more) store_stage(cur ^ 1);
        __syncthreads();
    }

    // epilogue: h (bf16).  acc[ct*4+rt] tile at rows rt*16, cols wave*64+ct*16
    const int colb = lane & 15;
    const int rowa = (lane >> 4) * 8;
    #pragma unroll
    for (int ct = 0; ct < 4; ++ct) {
        #pragma unroll
        for (int rt = 0; rt < 4; ++rt) {
            #pragma unroll
            for (int v = 0; v < 8; ++v) {
                int row = m0 + rt * 16 + v + rowa;
                int col = wave * 64 + ct * 16 + colb;
                hb[(size_t)row * F + col] = (__bf16)acc[ct * 4 + rt][v];
            }
        }
    }
}

// ------------------------------------- out = h @ W^T + b  (f32 out) --------
// Block: 256 threads = 8 waves over a 128x256 tile; wave = 4x4 sub-tiles
// (rows (w&1)*64, cols (w>>1)*64). K = 256, fully unrolled.
__global__ __launch_bounds__(256) void k_gemm_w(const __bf16* __restrict__ hb,
                                                const __bf16* __restrict__ Wb,
                                                const float*  __restrict__ bias,
                                                float*        __restrict__ out) {
    const int wave = threadIdx.x >> 5;
    const int lane = threadIdx.x & 31;
    const int mrow = blockIdx.x * 128 + (wave & 1) * 64;
    const int ncol = (wave >> 1) * 64;

    const int fr   = lane & 15;
    const int koff = (lane >> 4) * 8;

    v8f acc[16];
    #pragma unroll
    for (int i = 0; i < 16; ++i) acc[i] = (v8f){0.f,0.f,0.f,0.f,0.f,0.f,0.f,0.f};

    #pragma unroll
    for (int k0 = 0; k0 < F; k0 += KT) {
        v16bf a[4];
        #pragma unroll
        for (int rt = 0; rt < 4; ++rt)
            a[rt] = ld_frag(hb + (size_t)(mrow + rt * 16 + fr) * F + k0 + koff);
        #pragma unroll
        for (int ct = 0; ct < 4; ++ct) {
            // column n of W^T == row n of W (row-major bf16)
            v16bf b = ld_frag(Wb + (size_t)(ncol + ct * 16 + fr) * F + k0 + koff);
            #pragma unroll
            for (int rt = 0; rt < 4; ++rt)
                acc[ct * 4 + rt] = wmma_bf16(a[rt], b, acc[ct * 4 + rt]);
        }
    }

    const int colb = lane & 15;
    const int rowa = (lane >> 4) * 8;
    #pragma unroll
    for (int ct = 0; ct < 4; ++ct) {
        int col = ncol + ct * 16 + colb;
        float bv = bias[col];
        #pragma unroll
        for (int rt = 0; rt < 4; ++rt) {
            #pragma unroll
            for (int v = 0; v < 8; ++v) {
                int row = mrow + rt * 16 + v + rowa;
                out[(size_t)row * F + col] = acc[ct * 4 + rt][v] + bv;
            }
        }
    }
}

// ---------------------------------------------------------------------------
extern "C" void kernel_launch(void* const* d_in, const int* in_sizes, int n_in,
                              void* d_out, int out_size, void* d_ws, size_t ws_size,
                              hipStream_t stream) {
    const float* x   = (const float*)d_in[0];   // [8192,256]
    const float* adj = (const float*)d_in[1];   // [8192,8192]
    const float* W   = (const float*)d_in[2];   // [256,256]
    const float* b   = (const float*)d_in[3];   // [256]
    float*       out = (float*)d_out;           // [8192,256] f32

    char* ws = (char*)d_ws;
    float*  dis = (float*)ws;                             ws += (size_t)NN * 4;
    __bf16* xT  = (__bf16*)ws;                            ws += (size_t)F * NN * 2;
    __bf16* Wb  = (__bf16*)ws;                            ws += (size_t)F * F * 2;
    __bf16* hb  = (__bf16*)ws;

    k_rowsum  <<<NN, 256, 0, stream>>>(adj, dis);
    k_xT      <<<dim3(NN / 64, F / 64), 256, 0, stream>>>(x, xT);
    k_cvtW    <<<(F * F) / 256, 256, 0, stream>>>(W, Wb);
    k_gemm_adj<<<NN / MT, 128, 0, stream>>>(adj, dis, xT, hb);
    k_gemm_w  <<<NN / 128, 256, 0, stream>>>(hb, Wb, b, out);
}
